// LargeMambaModel_86208583565491
// MI455X (gfx1250) — matrix operand order repaired and checked by reading, not compiled, not run on hardware
//
#include <hip/hip_runtime.h>

// ---------------------------------------------------------------------------
// Types
// ---------------------------------------------------------------------------
typedef __attribute__((ext_vector_type(16))) __bf16 bf16x16;
typedef __attribute__((ext_vector_type(8)))  float  v8f;

#define L_DIM 2048
#define B_DIM 32
#define D_DIM 1024
#define M_DIM (L_DIM * B_DIM)   // 65536 rows
#define N_LAYERS 4

// bf16 <-> f32 helpers (round-to-nearest-even)
__device__ __forceinline__ __bf16 f2bf(float f) {
    unsigned u = __builtin_bit_cast(unsigned, f);
    unsigned r = u + 0x7FFFu + ((u >> 16) & 1u);
    unsigned short h = (unsigned short)(r >> 16);
    return __builtin_bit_cast(__bf16, h);
}
__device__ __forceinline__ float bf2f(__bf16 h) {
    unsigned short us = __builtin_bit_cast(unsigned short, h);
    unsigned u = ((unsigned)us) << 16;
    return __builtin_bit_cast(float, u);
}

// Branchless tanh: tanh(x) = sign(x) * (1 - e^-2|x|) / (1 + e^-2|x|)
__device__ __forceinline__ float fast_tanhf(float x) {
    float ax = __builtin_fabsf(x);
    float e  = __expf(-2.0f * ax);
    float t  = (1.0f - e) / (1.0f + e);
    return __builtin_copysignf(t, x);
}

// ---------------------------------------------------------------------------
// CDNA5 async global -> LDS copy (ASYNCcnt path).
// LDS operand: flat LDS-aperture addresses are {aperture_hi, lds_offset32},
// so the low 32 bits of a generic pointer to __shared__ are the DS address.
// ---------------------------------------------------------------------------
__device__ __forceinline__ void async_copy16(__bf16* lds_dst, const __bf16* gsrc) {
    unsigned lds = (unsigned)(unsigned long long)lds_dst;
    asm volatile("global_load_async_to_lds_b128 %0, %1, off"
                 :: "v"(lds), "v"(gsrc) : "memory");
}
#define WAIT_ASYNC(n) asm volatile("s_wait_asynccnt %0" :: "n"(n) : "memory")

// ---------------------------------------------------------------------------
// Kernel A: fp32 -> bf16 weight conversion (4 * 1024 * 1024 elements)
// ---------------------------------------------------------------------------
__global__ void convert_bf16_kernel(const float* __restrict__ in,
                                    __bf16* __restrict__ out, int n) {
    int i = (blockIdx.x * blockDim.x + threadIdx.x) * 4;
    if (i + 3 < n) {
        float4 v = *(const float4*)(in + i);
        out[i + 0] = f2bf(v.x);
        out[i + 1] = f2bf(v.y);
        out[i + 2] = f2bf(v.z);
        out[i + 3] = f2bf(v.w);
    }
}

// ---------------------------------------------------------------------------
// Kernel B: diagonal linear SSM scan  h[t] = a * h[t-1] + x[t]
// One thread per (b,d) channel; loads/stores coalesced over contiguous B*D.
// ---------------------------------------------------------------------------
__global__ void scan_kernel(const float* __restrict__ x,
                            const float* __restrict__ A_log,
                            __bf16* __restrict__ h) {
    const int BD  = B_DIM * D_DIM;                         // 32768
    const int idx = blockIdx.x * blockDim.x + threadIdx.x; // 0..BD-1
    const int d   = idx & (D_DIM - 1);
    const float a = __expf(A_log[d]);
    float hv = 0.0f;
    const float* xp = x + idx;
    __bf16*      hp = h + idx;
    for (int t = 0; t < L_DIM; ++t) {
        hv = fmaf(a, hv, xp[(size_t)t * BD]);
        hp[(size_t)t * BD] = f2bf(hv);
    }
}

// ---------------------------------------------------------------------------
// Kernel C: fused GEMM + bias + tanh, bf16 in / bf16 out, f32 accumulate.
//   Out[m,e] = tanh( sum_k A[m,k] * W[e,k] + bias[e] )
// Block tile 128(M) x 64(N), double-buffered K chunks of 64 staged with
// global_load_async_to_lds_b128. 8 wave32s: wave (wm,wn); each wave owns a
// 64x16 strip -> 8 x v_wmma_f32_16x16x32_bf16 per K chunk.
// ---------------------------------------------------------------------------
__global__ __launch_bounds__(256)
void gemm_tanh_kernel(const __bf16* __restrict__ A,
                      const __bf16* __restrict__ W,
                      const float*  __restrict__ bias,
                      __bf16* __restrict__ Out) {
    constexpr int K  = D_DIM;    // 1024
    constexpr int N  = D_DIM;    // 1024
    constexpr int BM = 128, BN = 64, BK = 64;
    constexpr int LDT = 72;      // padded LDS row stride (halves): 144B, 16B-aligned
    constexpr int TILE_LOADS = 6; // async b128 loads per thread per tile

    __shared__ __bf16 As[2][BM][LDT];   // 36 KB
    __shared__ __bf16 Ws[2][BN][LDT];   // 18 KB

    const int m0   = blockIdx.x * BM;
    const int n0   = blockIdx.y * BN;
    const int tid  = threadIdx.x;
    const int lane = tid & 31;
    const int wave = tid >> 5;
    const int wm   = wave & 1;          // 0..1 : which 64-row half
    const int wn   = wave >> 1;         // 0..3 : which 16-col strip

    v8f acc[4] = {v8f{}, v8f{}, v8f{}, v8f{}};

    // ---- async stage of one K tile (BK=64) into buffer `buf` ----
    auto stage = [&](int kc, int buf) {
#pragma unroll
        for (int s = 0; s < 4; ++s) {              // A: 128 rows x 8 chunks
            int c = tid + s * 256;
            int row = c >> 3;
            int off = (c & 7) * 8;
            async_copy16(&As[buf][row][off],
                         &A[(size_t)(m0 + row) * K + kc + off]);
        }
#pragma unroll
        for (int s = 0; s < 2; ++s) {              // W: 64 rows x 8 chunks
            int c = tid + s * 256;
            int row = c >> 3;
            int off = (c & 7) * 8;
            async_copy16(&Ws[buf][row][off],
                         &W[(size_t)(n0 + row) * K + kc + off]);
        }
    };

    stage(0, 0);                                    // prologue

    const int klo  = (lane >> 4) * 8;   // A frag: lane<16 -> K 0-7/16-23; else 8-15/24-31
    const int arow0 = wm * 64 + (lane & 15);
    const int bcol  = wn * 16 + (lane & 15);

    constexpr int NIT = K / BK;                     // 16 iterations
    for (int it = 0; it < NIT; ++it) {
        const int buf = it & 1;
        if (it + 1 < NIT) {
            stage((it + 1) * BK, buf ^ 1);          // prefetch next tile (async)
            WAIT_ASYNC(TILE_LOADS);                 // current tile landed (in-order)
        } else {
            WAIT_ASYNC(0);
        }
        __syncthreads();

#pragma unroll
        for (int kk = 0; kk < BK; kk += 32) {
            // B fragment (32x16): lane%16 = column, lane<16 -> K 0-7/16-23; else 8-15/24-31
            bf16x16 bfrag;
            {
                const __bf16* src = &Ws[buf][bcol][kk + (lane >> 4) * 8];
#pragma unroll
                for (int j = 0; j < 8; ++j) { bfrag[j] = src[j]; bfrag[j + 8] = src[16 + j]; }
            }
#pragma unroll
            for (int i = 0; i < 4; ++i) {
                const __bf16* arow = &As[buf][arow0 + i * 16][kk];
                bf16x16 afrag;
#pragma unroll
                for (int j = 0; j < 8; ++j) {
                    afrag[j]     = arow[klo + j];
                    afrag[j + 8] = arow[klo + 16 + j];
                }
                acc[i] = __builtin_amdgcn_wmma_f32_16x16x32_bf16(
                    false, afrag, false, bfrag, (short)0, acc[i], false, false);
            }
        }
        __syncthreads();   // all waves done reading buf before it is re-staged
    }

    // ---- epilogue: bias + tanh + bf16 store ----
    const int col   = n0 + bcol;
    const float bv  = bias[col];
    const int rhalf = (lane >> 4) * 8;   // C/D layout: VGPR j = rows j / j+8
#pragma unroll
    for (int i = 0; i < 4; ++i) {
        const int rbase = m0 + wm * 64 + i * 16 + rhalf;
#pragma unroll
        for (int j = 0; j < 8; ++j) {
            float v = fast_tanhf(acc[i][j] + bv);
            Out[(size_t)(rbase + j) * N + col] = f2bf(v);
        }
    }
}

// ---------------------------------------------------------------------------
// Kernel D: head projection  out[m] = dot(act[m,:], w) + b
// One wave32 per output row; shuffle reduction.
// ---------------------------------------------------------------------------
__global__ void head_kernel(const __bf16* __restrict__ act,
                            const float* __restrict__ w,
                            const float* __restrict__ b,
                            float* __restrict__ out) {
    const int wave = threadIdx.x >> 5;
    const int lane = threadIdx.x & 31;
    const int m    = blockIdx.x * 8 + wave;
    const __bf16* row = act + (size_t)m * D_DIM;
    float s = 0.0f;
    for (int d = lane; d < D_DIM; d += 32)
        s = fmaf(bf2f(row[d]), w[d], s);
#pragma unroll
    for (int off = 16; off > 0; off >>= 1)
        s += __shfl_down(s, off, 32);
    if (lane == 0) out[m] = s + b[0];
}

// ---------------------------------------------------------------------------
// Launch
// ---------------------------------------------------------------------------
extern "C" void kernel_launch(void* const* d_in, const int* in_sizes, int n_in,
                              void* d_out, int out_size, void* d_ws, size_t ws_size,
                              hipStream_t stream) {
    (void)in_sizes; (void)n_in; (void)out_size; (void)ws_size;

    const float* x        = (const float*)d_in[0];   // [L,B,D]
    const float* A_log    = (const float*)d_in[1];   // [D]
    const float* layers_W = (const float*)d_in[2];   // [4,D,D]
    const float* layers_b = (const float*)d_in[3];   // [4,D]
    const float* head_W   = (const float*)d_in[4];   // [1,D]
    const float* head_b   = (const float*)d_in[5];   // [1]
    float* out = (float*)d_out;                      // [L,B]

    // Workspace: two bf16 activation ping-pong buffers + bf16 weights
    const size_t actBytes = (size_t)M_DIM * D_DIM * sizeof(__bf16); // 128 MB
    char* ws = (char*)d_ws;
    __bf16* actA = (__bf16*)ws;
    __bf16* actB = (__bf16*)(ws + actBytes);
    __bf16* Wbf  = (__bf16*)(ws + 2 * actBytes);     // 8 MB

    // A: convert weights fp32 -> bf16
    {
        const int n = N_LAYERS * D_DIM * D_DIM;      // 4,194,304
        convert_bf16_kernel<<<n / (256 * 4), 256, 0, stream>>>(layers_W, Wbf, n);
    }

    // B: SSM scan -> actA (bf16)
    scan_kernel<<<(B_DIM * D_DIM) / 256, 256, 0, stream>>>(x, A_log, actA);

    // C: 4 fused GEMM+bias+tanh layers, ping-pong actA <-> actB
    dim3 ggrid(M_DIM / 128, D_DIM / 64);             // 512 x 16
    __bf16* cur = actA;
    __bf16* nxt = actB;
    for (int i = 0; i < N_LAYERS; ++i) {
        gemm_tanh_kernel<<<ggrid, 256, 0, stream>>>(
            cur, Wbf + (size_t)i * D_DIM * D_DIM, layers_b + i * D_DIM, nxt);
        __bf16* t = cur; cur = nxt; nxt = t;
    }
    // after 4 swaps, final activations are back in actA (== cur)

    // D: head projection -> out
    head_kernel<<<M_DIM / 8, 256, 0, stream>>>(cur, head_W, head_b, out);
}
